// AuxSeLoss_53498112639112
// MI455X (gfx1250) — compile-verified, matrix operand order checked by reading, not run on hardware
//
#include <hip/hip_runtime.h>
#include <hip/hip_bf16.h>

#define N_CLASSES 21
#define NBATCH 16
#define HW (256 * 256)
#define N_ELEM (NBATCH * N_CLASSES * HW)   // 22,020,096
#define N4 (N_ELEM / 4)                    // 5,505,024 float4s
#define BATCH_F4 ((N_CLASSES * HW) / 4)    // 344,064 float4s per batch
#define TPB 256
#define ITERS 8
#define TILE (TPB * ITERS)                 // 2048 float4s per block
#define NBLK ((N4 + TILE - 1) / TILE)      // 2688 blocks (exact, no tail)

typedef float    v2f  __attribute__((ext_vector_type(2)));
typedef float    v8f  __attribute__((ext_vector_type(8)));
typedef _Float16 v16h __attribute__((ext_vector_type(16)));

// Numerically stable BCE-with-logits term:
// max(x,0) - x*t + log(1 + exp(-|x|)).  Log arg is in (1,2] -> fast log is safe.
__device__ __forceinline__ float bce_term(float x, float t) {
    float ax = __builtin_fabsf(x);
    return fmaxf(x, 0.0f) - x * t + __logf(1.0f + __expf(-ax));
}

// torch.histc-equivalent bin bit: idx = clip(floor(x * n/(n-1)), 0, n-1), valid for 0<=x<=n-1
__device__ __forceinline__ unsigned class_bit(float x) {
    if (x < 0.0f || x > (float)(N_CLASSES - 1)) return 0u;
    int idx = (int)floorf(x * ((float)N_CLASSES / (float)(N_CLASSES - 1)));
    idx = idx > (N_CLASSES - 1) ? (N_CLASSES - 1) : idx;
    idx = idx < 0 ? 0 : idx;
    return 1u << idx;
}

// Sum 256 f32 partials staged in LDS using WMMA as a reduction engine.
// With A = all-ones, D[i][j] = sum_k B[k][j]; every B element contributes to
// exactly one column sum, so row 0 of D holds 16 disjoint partial sums and the
// result is independent of the exact lane->B mapping. Must be called by a
// fully-active wave (EXEC all 1s). Returns total in all lanes of group 0..15.
__device__ __forceinline__ float wmma_reduce256(const float* red, int lane) {
#if __has_builtin(__builtin_amdgcn_wmma_f32_16x16x4_f32)
    v8f acc = {};
    v2f ones; ones.x = 1.0f; ones.y = 1.0f;
#pragma unroll
    for (int c = 0; c < 4; ++c) {
        v2f b;
        b.x = red[c * 64 + lane];
        b.y = red[c * 64 + 32 + lane];
        acc = __builtin_amdgcn_wmma_f32_16x16x4_f32(
            false, ones, false, b, (short)0, acc, false, false);
    }
    float v = acc[0];   // D[0][j] in lanes 0..15 (dup in 16..31)
#else
    // Fallback: confirmed f16 WMMA; 256 partials in half of B, rest zero.
    v16h a, b;
#pragma unroll
    for (int i = 0; i < 16; ++i) a[i] = (_Float16)1.0f;
#pragma unroll
    for (int i = 0; i < 8; ++i)  b[i] = (_Float16)red[i * 32 + lane];
#pragma unroll
    for (int i = 8; i < 16; ++i) b[i] = (_Float16)0.0f;
    v8f acc = {};
    acc = __builtin_amdgcn_wmma_f32_16x16x32_f16(
        false, a, false, b, (short)0, acc, false, false);
    float v = acc[0];
#endif
    // Sum the 16 column sums (lanes 16..31 are duplicates of 0..15).
    v += __shfl_xor(v, 1);
    v += __shfl_xor(v, 2);
    v += __shfl_xor(v, 4);
    v += __shfl_xor(v, 8);
    return v;
}

__global__ __launch_bounds__(32) void auxse_init(double* gsum, unsigned* gmask) {
    int t = threadIdx.x;
    if (t < 2) gsum[t] = 0.0;
    if (t < NBATCH) gmask[t] = 0u;
}

__global__ __launch_bounds__(TPB) void auxse_main(
    const float* __restrict__ out0, const float* __restrict__ out1,
    const float* __restrict__ tgt, double* __restrict__ gsum,
    unsigned* __restrict__ gmask) {
    __shared__ float    red0[TPB];
    __shared__ float    red1[TPB];
    __shared__ unsigned lmask[NBATCH];

    const int tid = threadIdx.x;
    if (tid < NBATCH) lmask[tid] = 0u;
    __syncthreads();

    const float4* o0 = (const float4*)out0;
    const float4* o1 = (const float4*)out1;
    const float4* tg = (const float4*)tgt;

    const long long base = (long long)blockIdx.x * TILE;
    float s0 = 0.0f, s1 = 0.0f;
    unsigned curb = 0xFFFFFFFFu, curm = 0u;

#pragma unroll
    for (int it = 0; it < ITERS; ++it) {
        long long i = base + (long long)it * TPB + tid;
        if (i < N4) {
            unsigned b = (unsigned)(i / BATCH_F4);
            if (b != curb) {
                if (curm) atomicOr(&lmask[curb], curm);
                curb = b; curm = 0u;
            }
            float4 x0 = o0[i];
            float4 x1 = o1[i];
            float4 t  = tg[i];
            s0 += bce_term(x0.x, t.x) + bce_term(x0.y, t.y)
                + bce_term(x0.z, t.z) + bce_term(x0.w, t.w);
            s1 += bce_term(x1.x, t.x) + bce_term(x1.y, t.y)
                + bce_term(x1.z, t.z) + bce_term(x1.w, t.w);
            curm |= class_bit(t.x) | class_bit(t.y)
                  | class_bit(t.z) | class_bit(t.w);
        }
    }
    if (curm && curb != 0xFFFFFFFFu) atomicOr(&lmask[curb], curm);

    red0[tid] = s0;
    red1[tid] = s1;
    __syncthreads();

    if (tid < 32) {  // one fully-active wave: EXEC all 1s for WMMA
        float t0 = wmma_reduce256(red0, tid);
        float t1 = wmma_reduce256(red1, tid);
        if (tid == 0) {
            atomicAdd(&gsum[0], (double)t0);
            atomicAdd(&gsum[1], (double)t1);
        }
    }
    __syncthreads();
    if (tid < NBATCH) {
        unsigned m = lmask[tid];
        if (m) atomicOr(&gmask[tid], m);
    }
}

__global__ __launch_bounds__(352) void auxse_final(
    const float* __restrict__ out2, const double* __restrict__ gsum,
    const unsigned* __restrict__ gmask, float* __restrict__ out) {
    __shared__ float red[352];
    const int tid = threadIdx.x;
    float v = 0.0f;
    if (tid < NBATCH * N_CLASSES) {
        int b = tid / N_CLASSES;
        int c = tid % N_CLASSES;
        float t = (float)((gmask[b] >> c) & 1u);
        v = bce_term(out2[tid], t);
    }
    red[tid] = v;
    __syncthreads();
    if (tid == 0) {
        float se = 0.0f;
        for (int i = 0; i < 352; ++i) se += red[i];
        double l0  = gsum[0] / (double)N_ELEM;
        double l1  = gsum[1] / (double)N_ELEM;
        double lse = (double)se / (double)(NBATCH * N_CLASSES);
        out[0] = (float)(l0 + 0.4 * l1 + 0.2 * lse);
    }
}

extern "C" void kernel_launch(void* const* d_in, const int* in_sizes, int n_in,
                              void* d_out, int out_size, void* d_ws, size_t ws_size,
                              hipStream_t stream) {
    (void)in_sizes; (void)n_in; (void)out_size; (void)ws_size;
    const float* out0 = (const float*)d_in[0];
    const float* out1 = (const float*)d_in[1];
    const float* out2 = (const float*)d_in[2];
    const float* tgt  = (const float*)d_in[3];

    double*   gsum  = (double*)d_ws;                       // 2 x f64 at offset 0
    unsigned* gmask = (unsigned*)((char*)d_ws + 16);       // 16 x u32

    auxse_init<<<1, 32, 0, stream>>>(gsum, gmask);
    auxse_main<<<NBLK, TPB, 0, stream>>>(out0, out1, tgt, gsum, gmask);
    auxse_final<<<1, 352, 0, stream>>>(out2, gsum, gmask, (float*)d_out);
}
